// VectorAttention_42631845380169
// MI455X (gfx1250) — compile-verified, matrix operand order checked by reading, not run on hardware
//
#include <hip/hip_runtime.h>
#include <math.h>

// ---------------------------------------------------------------------------
// Problem constants (match reference)
// ---------------------------------------------------------------------------
#define B_ 2
#define N_ 4096
#define C_ 256
#define K_ 32
#define G_ 32
#define H_ 8
#define CHUNK 512          // KNN row-chunk processed per gram/topk pass

typedef _Float16 half_t;
typedef __attribute__((ext_vector_type(16))) _Float16 v16h;
typedef __attribute__((ext_vector_type(8)))  _Float16 v8h;
typedef __attribute__((ext_vector_type(8)))  float    v8f;

// ---------------------------------------------------------------------------
// WMMA helpers
// ---------------------------------------------------------------------------
__device__ __forceinline__ v8f wmma_f16(v16h a, v16h b, v8f c) {
  // D = A(16x32,f16) * B(32x16,f16) + C(16x16,f32)
  return __builtin_amdgcn_wmma_f32_16x16x32_f16(
      /*neg_a=*/false, a, /*neg_b=*/false, b,
      /*c_mod=*/(short)0, c, /*reuse_a=*/false, /*reuse_b=*/false);
}

// A-operand per-lane load from a row-major f16 matrix (LDS or global).
// Lane l (0..31): row = l%16 (+16*mt applied by caller), K-chunks at
// c0 = ks*32 + (l/16)*8 and c0+16 -> matches ISA 16-bit A 16x32 layout.
__device__ __forceinline__ v16h load_tile_rows(const half_t* base, int pitch,
                                               int row, int c0) {
  v16h a;
  ((v8h*)&a)[0] = *(const v8h*)(base + (size_t)row * pitch + c0);
  ((v8h*)&a)[1] = *(const v8h*)(base + (size_t)row * pitch + c0 + 16);
  return a;
}

// B-operand load from the pre-swizzled weight buffer: tile-major, 32 lanes x
// 16 contiguous f16 each -> two global_load_b128 per lane per tile.
__device__ __forceinline__ v16h load_b_swz(const half_t* w, int tile, int lane) {
  const half_t* p = w + ((size_t)tile * 32 + lane) * 16;
  v16h b;
  ((v8h*)&b)[0] = *(const v8h*)(p);
  ((v8h*)&b)[1] = *(const v8h*)(p + 8);
  return b;
}

__device__ __forceinline__ float leaky(float y) { return y >= 0.f ? y : 0.2f * y; }

// ---------------------------------------------------------------------------
// Prep: fold eval-BN into weights, convert to f16, swizzle into B layout.
//   scale[o] = gamma/sqrt(var+eps);  W'[o][c] = scale*W;  shift[o] folded.
// Swizzle index: ((kt*NT + nt)*32 + lane)*16 + e ; e<8 -> c0+e, else c0+16+(e-8)
// ---------------------------------------------------------------------------
__global__ void prep_weights_kernel(const float* __restrict__ W,
                                    const float* __restrict__ bias,
                                    const float* __restrict__ gamma,
                                    const float* __restrict__ beta,
                                    const float* __restrict__ mean,
                                    const float* __restrict__ var,
                                    half_t* __restrict__ Wswz,
                                    float* __restrict__ shift,
                                    int Cout, int Cin) {
  int tid = blockIdx.x * blockDim.x + threadIdx.x;
  int NT = Cout >> 4;
  int total = Cout * Cin;
  if (tid < total) {
    int t = tid >> 9;          // tile (512 elements each)
    int r = tid & 511;
    int l = r >> 4;            // lane
    int e = r & 15;            // element within lane
    int nt = t % NT;
    int kt = t / NT;
    int o = nt * 16 + (l & 15);
    int c = kt * 32 + ((l >> 4) * 8) + (e < 8 ? e : e + 8);
    float scale = gamma[o] * rsqrtf(var[o] + 1e-5f);
    Wswz[tid] = (half_t)(scale * W[(size_t)o * Cin + c]);
  }
  if (tid < Cout) {
    float scale = gamma[tid] * rsqrtf(var[tid] + 1e-5f);
    shift[tid] = (bias[tid] - mean[tid]) * scale + beta[tid];
  }
}

__global__ void f32_to_f16_kernel(const float* __restrict__ src,
                                  half_t* __restrict__ dst, int n) {
  int tid = blockIdx.x * blockDim.x + threadIdx.x;
  if (tid < n) dst[tid] = (half_t)src[tid];
}

// ---------------------------------------------------------------------------
// Projection: Y = leaky(X @ W'^T + shift). 64 rows x 256 cols per workgroup,
// 8 waves * 8 tiles, A staged through LDS, B streamed from swizzled weights.
// ---------------------------------------------------------------------------
__global__ __launch_bounds__(256) void proj_kernel(
    const half_t* __restrict__ Xh, const half_t* __restrict__ Wswz,
    const float* __restrict__ shift, float* __restrict__ Yf,
    half_t* __restrict__ Yh, int writeHalf) {
  __shared__ half_t xs[64 * C_];
  int tid = threadIdx.x;
  int row0 = blockIdx.x * 64;

  // Stage 64 contiguous rows of X into LDS (2048 x 16B chunks).
  {
    const v8h* src = (const v8h*)(Xh + (size_t)row0 * C_);
    v8h* dst = (v8h*)xs;
    for (int i = 0; i < 8; ++i) dst[tid + 256 * i] = src[tid + 256 * i];
  }
  __syncthreads();

  int w = tid >> 5, l = tid & 31;
  for (int i = 0; i < 8; ++i) {
    int tile = w + 8 * i;            // 64 tiles: 4 mt x 16 nt
    int mt = tile >> 4, nt = tile & 15;
    v8f acc = {};
    for (int kt = 0; kt < 8; ++kt) {
      __builtin_prefetch(Wswz + ((size_t)((kt + 1) * 16 + nt) * 32 + l) * 16, 0, 1);
      v16h a = load_tile_rows(xs, C_, mt * 16 + (l & 15), kt * 32 + ((l >> 4) * 8));
      v16h b = load_b_swz(Wswz, kt * 16 + nt, l);
      acc = wmma_f16(a, b, acc);
    }
    int col = nt * 16 + (l & 15);
    float sh = shift[col];
    for (int v = 0; v < 8; ++v) {
      int row = row0 + mt * 16 + v + 8 * (l >> 4);
      float y = leaky(acc[v] + sh);
      Yf[(size_t)row * C_ + col] = y;
      if (writeHalf) Yh[(size_t)row * C_ + col] = (half_t)y;
    }
  }
}

// ---------------------------------------------------------------------------
// Row squared norms (f16 in, f32 out) -- needed for KNN score 2*FF^T - sq[j].
// ---------------------------------------------------------------------------
__global__ void sqnorm_kernel(const half_t* __restrict__ Fh,
                              float* __restrict__ sq, int rows) {
  int r = blockIdx.x * blockDim.x + threadIdx.x;
  if (r >= rows) return;
  const half_t* p = Fh + (size_t)r * C_;
  float s = 0.f;
  for (int c = 0; c < C_; ++c) { float f = (float)p[c]; s += f * f; }
  sq[r] = s;
}

// ---------------------------------------------------------------------------
// Gram chunk: D[i-i0][j] = 2*dot(f_i,f_j) - sq[j] for a CHUNK x N_ slab.
// One 16x16 WMMA tile per wave; -sq_i dropped (row-constant, rank-invariant).
// ---------------------------------------------------------------------------
__global__ __launch_bounds__(256) void gram_kernel(
    const half_t* __restrict__ Fh, const float* __restrict__ sq,
    float* __restrict__ D, int i0) {
  int w = threadIdx.x >> 5, l = threadIdx.x & 31;
  int tile = blockIdx.x * 8 + w;        // (CHUNK/16) x (N_/16) tiles
  int mt = tile >> 8;                   // N_/16 == 256
  int nt = tile & 255;
  int arow = i0 + mt * 16 + (l & 15);
  int brow = nt * 16 + (l & 15);
  int coff = (l >> 4) * 8;
  v8f acc = {};
  for (int kt = 0; kt < 8; ++kt) {
    v16h a = load_tile_rows(Fh, C_, arow, kt * 32 + coff);
    v16h b = load_tile_rows(Fh, C_, brow, kt * 32 + coff);
    acc = wmma_f16(a, b, acc);
  }
  int j = nt * 16 + (l & 15);
  float sj = sq[j];
  for (int v = 0; v < 8; ++v) {
    int i = mt * 16 + v + 8 * (l >> 4);
    D[(size_t)i * N_ + j] = 2.f * acc[v] - sj;
  }
}

// ---------------------------------------------------------------------------
// Row-wise top-32 selection (descending, stable on ties) over a D chunk.
// ---------------------------------------------------------------------------
__global__ void topk_kernel(const float* __restrict__ D, int* __restrict__ idx_out) {
  int r = blockIdx.x * blockDim.x + threadIdx.x;
  if (r >= CHUNK) return;
  float bd[K_]; int bi[K_];
  for (int k = 0; k < K_; ++k) { bd[k] = -3.4e38f; bi[k] = 0; }
  const float* row = D + (size_t)r * N_;
  for (int j = 0; j < N_; ++j) {
    float d = row[j];
    if (d > bd[K_ - 1]) {
      int p = K_ - 1;
      while (p > 0 && bd[p - 1] < d) { bd[p] = bd[p - 1]; bi[p] = bi[p - 1]; --p; }
      bd[p] = d; bi[p] = j;
    }
  }
  for (int k = 0; k < K_; ++k) idx_out[(size_t)r * K_ + k] = bi[k];
}

// ---------------------------------------------------------------------------
// Fused per-point vector attention. One workgroup = one (b,n) point.
// LDS plan (53248 B dynamic):
//   qld(1K) shp(1K) sh1/sh2/nbrk/nbrv(4x128)
//   vals  f16 32x264  (vf gather + posb)
//   vecs  f16 32x264  (key - q + posb)
//   Amp   f16 32x264  (delta-pos A tile; region reused for e1/e2/ps later)
// ---------------------------------------------------------------------------
__global__ __launch_bounds__(256) void attn_kernel(
    const float* __restrict__ pos, const float* __restrict__ qf,
    const float* __restrict__ kf, const float* __restrict__ vf,
    const int* __restrict__ idxk, const int* __restrict__ idxv,
    const half_t* __restrict__ Wp, const half_t* __restrict__ We1,
    const half_t* __restrict__ We2, const float* __restrict__ shp_g,
    const float* __restrict__ sh1_g, const float* __restrict__ sh2_g,
    float* __restrict__ out) {
  extern __shared__ unsigned char smem[];
  float*  qld  = (float*)(smem + 0);
  float*  shp  = (float*)(smem + 1024);
  float*  sh1  = (float*)(smem + 2048);
  float*  sh2  = (float*)(smem + 2176);
  int*    nbrk = (int*)  (smem + 2304);
  int*    nbrv = (int*)  (smem + 2432);
  half_t* vals = (half_t*)(smem + 2560);                 // 32 x 264 f16
  half_t* vecs = (half_t*)(smem + 19456);                // 32 x 264 f16
  half_t* Amp  = (half_t*)(smem + 36352);                // 32 x 264 f16
  half_t* e1   = (half_t*)(smem + 36352);                // overlays dead Amp
  float*  e2   = (float*)(smem + 36352 + 2560);          // 32 x 36 f32
  float*  ps   = (float*)(smem + 36352 + 2560 + 4608);   // 32 x 36 f32

  int tid = threadIdx.x;
  int w = tid >> 5, l = tid & 31;
  int pt = blockIdx.x;                        // b*N + n
  size_t baseC = (size_t)pt * C_;
  size_t bN = (size_t)(pt / N_) * N_;

  // Stage 1: point-local constants into LDS.
  qld[tid] = qf[baseC + tid];
  shp[tid] = shp_g[tid];
  if (tid < 32) {
    sh1[tid]  = sh1_g[tid];
    sh2[tid]  = sh2_g[tid];
    nbrk[tid] = idxk[(size_t)pt * K_ + tid];
    nbrv[tid] = idxv[(size_t)pt * K_ + tid];
  }
  __syncthreads();

  // Stage 2: A tile = (pos[neighbor] - pos[center]) as f16, 32 x 256.
  {
    float pc = pos[baseC + tid];
    for (int r = 0; r < K_; ++r) {
      float pn = pos[(bN + (size_t)nbrk[r]) * C_ + tid];
      Amp[r * 264 + tid] = (half_t)(pn - pc);
    }
  }
  __syncthreads();

  // Stage 3: posb = leaky(mpos @ Wp'^T + shp); fuse key/value gathers.
  for (int i = 0; i < 4; ++i) {
    int tile = w + 8 * i;                  // 32 tiles: 2 mt x 16 nt
    int mt = tile >> 4, nt = tile & 15;
    v8f acc = {};
    for (int kt = 0; kt < 8; ++kt) {
      v16h a = load_tile_rows(Amp, 264, mt * 16 + (l & 15), kt * 32 + ((l >> 4) * 8));
      v16h b = load_b_swz(Wp, kt * 16 + nt, l);
      acc = wmma_f16(a, b, acc);
    }
    int col = nt * 16 + (l & 15);
    float sh = shp[col];
    float qc = qld[col];
    for (int v = 0; v < 8; ++v) {
      int row = mt * 16 + v + 8 * (l >> 4);        // neighbor slot 0..31
      float y  = leaky(acc[v] + sh);
      float kg = kf[(bN + (size_t)nbrk[row]) * C_ + col];
      float vg = vf[(bN + (size_t)nbrv[row]) * C_ + col];
      vecs[row * 264 + col] = (half_t)(kg - qc + y);
      vals[row * 264 + col] = (half_t)(vg + y);
    }
  }
  __syncthreads();

  // Stage 4: enc1 = leaky(vecs(32x256) @ We1'^T(256x32) + sh1).
  if (w < 4) {
    int mt = w >> 1, nt = w & 1;
    v8f acc = {};
    for (int kt = 0; kt < 8; ++kt) {
      v16h a = load_tile_rows(vecs, 264, mt * 16 + (l & 15), kt * 32 + ((l >> 4) * 8));
      v16h b = load_b_swz(We1, kt * 2 + nt, l);
      acc = wmma_f16(a, b, acc);
    }
    int col = nt * 16 + (l & 15);
    float sh = sh1[col];
    for (int v = 0; v < 8; ++v) {
      int row = mt * 16 + v + 8 * (l >> 4);
      e1[row * 40 + col] = (half_t)leaky(acc[v] + sh);
    }
  }
  __syncthreads();

  // Stage 5: enc2 = leaky(e1(32x32) @ We2'^T(32x32) + sh2).
  if (w < 4) {
    int mt = w >> 1, nt = w & 1;
    v16h a = load_tile_rows(e1, 40, mt * 16 + (l & 15), (l >> 4) * 8);
    v16h b = load_b_swz(We2, nt, l);
    v8f acc = {};
    acc = wmma_f16(a, b, acc);
    int col = nt * 16 + (l & 15);
    float sh = sh2[col];
    for (int v = 0; v < 8; ++v) {
      int row = mt * 16 + v + 8 * (l >> 4);
      e2[row * 36 + col] = leaky(acc[v] + sh);
    }
  }
  __syncthreads();

  // Stage 6: softmax over the K=32 neighbor axis, one lane per g.
  if (tid < G_) {
    float m = -3.4e38f;
    for (int k = 0; k < K_; ++k) m = fmaxf(m, e2[k * 36 + tid]);
    float s = 0.f;
    for (int k = 0; k < K_; ++k) {
      float e = __expf(e2[k * 36 + tid] - m);
      ps[k * 36 + tid] = e;
      s += e;
    }
    float inv = 1.f / s;
    for (int k = 0; k < K_; ++k) ps[k * 36 + tid] *= inv;
  }
  __syncthreads();

  // Stage 7: out[c] = sum_k vals[k][c] * p[k][c/H]  (c = g*H + h).
  {
    int g = tid >> 3;
    float acc = 0.f;
    for (int k = 0; k < K_; ++k)
      acc += (float)vals[k * 264 + tid] * ps[k * 36 + g];
    out[baseC + tid] = acc;
  }
}

// ---------------------------------------------------------------------------
// Host orchestration
// ---------------------------------------------------------------------------
extern "C" void kernel_launch(void* const* d_in, const int* in_sizes, int n_in,
                              void* d_out, int out_size, void* d_ws, size_t ws_size,
                              hipStream_t stream) {
  (void)in_sizes; (void)n_in; (void)out_size; (void)ws_size;

  const float* x   = (const float*)d_in[0];
  const float* pos = (const float*)d_in[1];
  // params flattened in dict order: q,k,v,pos_bias,enc1,enc2 x {W,b,gamma,beta,mean,var}
  const float* P[6][6];
  for (int blk = 0; blk < 6; ++blk)
    for (int f = 0; f < 6; ++f)
      P[blk][f] = (const float*)d_in[2 + blk * 6 + f];

  const int BN  = B_ * N_;
  const int BNC = BN * C_;

  // Workspace carve (256B aligned).
  unsigned char* ws = (unsigned char*)d_ws;
  size_t off = 0;
  auto carve = [&](size_t bytes) -> unsigned char* {
    unsigned char* p = ws + off;
    off = (off + bytes + 255) & ~(size_t)255;
    return p;
  };
  half_t* xb  = (half_t*)carve((size_t)BNC * 2);
  half_t* wq  = (half_t*)carve((size_t)C_ * C_ * 2);
  half_t* wk  = (half_t*)carve((size_t)C_ * C_ * 2);
  half_t* wv  = (half_t*)carve((size_t)C_ * C_ * 2);
  half_t* wp  = (half_t*)carve((size_t)C_ * C_ * 2);
  half_t* we1 = (half_t*)carve((size_t)G_ * C_ * 2);
  half_t* we2 = (half_t*)carve((size_t)G_ * G_ * 2);
  float*  shq = (float*)carve(C_ * 4);
  float*  shk = (float*)carve(C_ * 4);
  float*  shv = (float*)carve(C_ * 4);
  float*  shp = (float*)carve(C_ * 4);
  float*  sh1 = (float*)carve(G_ * 4);
  float*  sh2 = (float*)carve(G_ * 4);
  float*  qf  = (float*)carve((size_t)BNC * 4);
  float*  kfb = (float*)carve((size_t)BNC * 4);
  float*  vfb = (float*)carve((size_t)BNC * 4);
  half_t* kh  = (half_t*)carve((size_t)BNC * 2);
  half_t* vh  = (half_t*)carve((size_t)BNC * 2);
  float*  sqk = (float*)carve((size_t)BN * 4);
  float*  sqv = (float*)carve((size_t)BN * 4);
  int*    idxk = (int*)carve((size_t)BN * K_ * 4);
  int*    idxv = (int*)carve((size_t)BN * K_ * 4);
  float*  Dch = (float*)carve((size_t)CHUNK * N_ * 4);

  // 1) Fold BN, swizzle weights to WMMA-B layout, compute shifts.
  {
    int couts[6]  = {C_, C_, C_, C_, G_, G_};
    int cins[6]   = {C_, C_, C_, C_, C_, G_};
    half_t* Wsz[6]  = {wq, wk, wv, wp, we1, we2};
    float*  Shf[6]  = {shq, shk, shv, shp, sh1, sh2};
    for (int blk = 0; blk < 6; ++blk) {
      int total = couts[blk] * cins[blk];
      prep_weights_kernel<<<(total + 255) / 256, 256, 0, stream>>>(
          P[blk][0], P[blk][1], P[blk][2], P[blk][3], P[blk][4], P[blk][5],
          Wsz[blk], Shf[blk], couts[blk], cins[blk]);
    }
  }

  // 2) x -> f16.
  f32_to_f16_kernel<<<(BNC + 255) / 256, 256, 0, stream>>>(x, xb, BNC);

  // 3) q/k/v projections (WMMA GEMM, BN+LeakyReLU fused).
  proj_kernel<<<BN / 64, 256, 0, stream>>>(xb, wq, shq, qf, (half_t*)nullptr, 0);
  proj_kernel<<<BN / 64, 256, 0, stream>>>(xb, wk, shk, kfb, kh, 1);
  proj_kernel<<<BN / 64, 256, 0, stream>>>(xb, wv, shv, vfb, vh, 1);

  // 4) Row norms for KNN scoring.
  sqnorm_kernel<<<BN / 256, 256, 0, stream>>>(kh, sqk, BN);
  sqnorm_kernel<<<BN / 256, 256, 0, stream>>>(vh, sqv, BN);

  // 5) Feature-space KNN: WMMA gram chunks + scalar top-32.
  for (int b = 0; b < B_; ++b) {
    for (int f = 0; f < 2; ++f) {
      const half_t* Fh = (f == 0 ? kh : vh) + (size_t)b * N_ * C_;
      const float*  sq = (f == 0 ? sqk : sqv) + (size_t)b * N_;
      int* idx = (f == 0 ? idxk : idxv);
      for (int i0 = 0; i0 < N_; i0 += CHUNK) {
        gram_kernel<<<((CHUNK / 16) * (N_ / 16)) / 8, 256, 0, stream>>>(Fh, sq, Dch, i0);
        topk_kernel<<<CHUNK / 128, 128, 0, stream>>>(Dch, idx + ((size_t)b * N_ + i0) * K_);
      }
    }
  }

  // 6) Fused per-point attention (pos_bias GEMM + enc1/enc2 + softmax + reduce).
  attn_kernel<<<BN, 256, 53248, stream>>>(pos, qf, kfb, vfb, idxk, idxv,
                                          wp, we1, we2, shp, sh1, sh2,
                                          (float*)d_out);
}
